// VectorQuantizer_34651796144744
// MI455X (gfx1250) — matrix operand order repaired and checked by reading, not compiled, not run on hardware
//
#include <hip/hip_runtime.h>
#include <hip/hip_bf16.h>
#include <math.h>

// ---------------------------------------------------------------------------
// VQ-VAE vector quantizer, MI455X (gfx1250, wave32, WMMA bf16 16x16x32)
//
// Outputs (flat, in return order):
//   d_out[0]                      loss (f32 scalar)
//   d_out[1 .. 16777216]          out  [64,256,32,32]  (= emb[idx] transposed)
//   d_out[16777217]               perplexity
//   d_out[16777218 .. +67108863]  encodings [65536,1024] one-hot
// ---------------------------------------------------------------------------

#define NUM_EMB   1024
#define EMB_DIM   256
#define NROWS     65536          // 64*32*32
#define HW        1024           // 32*32
#define OUT_ELEMS 16777216       // 64*256*1024
#define ENC_OFF   16777218       // 1 + OUT_ELEMS + 1
#define PERP_OFF  16777217

typedef __attribute__((ext_vector_type(16))) __bf16 v16bf;
typedef __attribute__((ext_vector_type(8)))  float  v8f;

union AFrag { unsigned short u[16]; v16bf v; };

__device__ __forceinline__ unsigned short f2bf(float f) {
    unsigned int u = __float_as_uint(f);
    u = (u + 0x7FFFu + ((u >> 16) & 1u)) >> 16;   // round-to-nearest-even
    return (unsigned short)u;
}
__device__ __forceinline__ float bf2f(unsigned short h) {
    return __uint_as_float(((unsigned int)h) << 16);
}

// ---------------------------------------------------------------------------
// prep: zero the histogram (ws is poisoned, and main increments it each call)
// ---------------------------------------------------------------------------
__global__ void vq_init(int* __restrict__ counts) {
    for (int k = threadIdx.x; k < NUM_EMB; k += 256) counts[k] = 0;
}

// ---------------------------------------------------------------------------
// prep: pre-swizzle emb into WMMA B-fragment layout (hi/lo bf16 split).
// B is 32x16 (K=c-chunk of 32, N=16 codes). B layout used here (contiguous k order, same as A):
//   lane L: n = L%16, c_local = (L/16)*16 + e   (e = 0..15 contiguous per lane)
// fragment unit = 16 bf16 (32B) per (k_tile, chunk, lane).
// ---------------------------------------------------------------------------
__global__ __launch_bounds__(256) void vq_prep_emb(
        const float* __restrict__ emb,
        unsigned short* __restrict__ fragH,
        unsigned short* __restrict__ fragL) {
    const int kt = blockIdx.x;            // 0..63
    const int t  = threadIdx.x;
    const int ch = t >> 5;                // 0..7 (c-chunk of 32)
    const int ln = t & 31;                // fragment lane
    const int k  = kt * 16 + (ln & 15);
    const size_t base = (((size_t)kt * 8 + ch) * 32 + ln) * 16;
#pragma unroll
    for (int e = 0; e < 16; ++e) {
        const int c = ch * 32 + ((ln >> 4) << 4) + e;
        const float x = emb[(size_t)k * EMB_DIM + c];
        const unsigned short h = f2bf(x);
        fragH[base + e] = h;
        fragL[base + e] = f2bf(x - bf2f(h));
    }
}

// prep: ||e_k||^2
__global__ void vq_enorm(const float* __restrict__ emb, float* __restrict__ enorm2) {
    const int k = blockIdx.x * 256 + threadIdx.x;
    if (k < NUM_EMB) {
        float s = 0.f;
#pragma unroll 8
        for (int c = 0; c < EMB_DIM; ++c) {
            const float v = emb[(size_t)k * EMB_DIM + c];
            s += v * v;
        }
        enorm2[k] = s;
    }
}

// ---------------------------------------------------------------------------
// main fused kernel: 512 workgroups x 256 threads (8 waves).
// Each wave owns 16 rows of x (A tile in VGPRs, bf16 hi/lo, contiguous k order),
// scans all 64 k-tiles with 3 WMMAs per 32-wide c-chunk (bf16x3 split),
// tracks running argmin, then the workgroup emits out / loss / one-hot /
// histogram for its 128 rows.
// ---------------------------------------------------------------------------
__global__ __launch_bounds__(256) void vq_main(
        const float* __restrict__ inp,
        const float* __restrict__ emb,
        const unsigned short* __restrict__ fragH,
        const unsigned short* __restrict__ fragL,
        const float* __restrict__ enorm2,
        int*   __restrict__ counts,
        float* __restrict__ partials,
        float* __restrict__ dout) {

    __shared__ __align__(32) unsigned short Blds[2][8][32][16]; // 16 KB
    __shared__ float enormLds[NUM_EMB];
    __shared__ int   idxLds[128];
    __shared__ float red[256];

    const int tid  = threadIdx.x;
    const int wave = tid >> 5;
    const int lane = tid & 31;
    const int wg   = blockIdx.x;
    const int n0   = wg * 128;            // first flat row of this WG
    const int b    = n0 >> 10;            // batch (HW = 1024 rows per batch)
    const int hw0  = n0 & (HW - 1);

    for (int k = tid; k < NUM_EMB; k += 256) enormLds[k] = enorm2[k];

    // ---- build A fragments (hi/lo) in registers -------------------------
    // 16-bit A 16x32 layout (matches B fragments): lane L: m = L%16,
    //   c_local(e) = ((L>>4)&1)*16 + e
    AFrag aH[8], aL[8];
    {
        const int m = lane & 15;
        const float* xcol = inp + (size_t)b * EMB_DIM * HW + hw0 + wave * 16 + m;
        const int grp8 = ((lane >> 4) & 1) * 8;
#pragma unroll
        for (int ch = 0; ch < 8; ++ch) {
#pragma unroll
            for (int e = 0; e < 16; ++e) {
                const int c = ch * 32 + (((lane >> 4) & 1) << 4) + e;
                const float x = xcol[(size_t)c * HW];
                const unsigned short h = f2bf(x);
                aH[ch].u[e] = h;
                aL[ch].u[e] = f2bf(x - bf2f(h));
            }
        }
    }
    __syncthreads();

    // ---- scan all 64 k-tiles -------------------------------------------
    float bv[8];
    int   bi[8];
#pragma unroll
    for (int i = 0; i < 8; ++i) { bv[i] = 3.0e38f; bi[i] = 0; }

    for (int kt = 0; kt < 64; ++kt) {
        // stage B fragments (hi+lo, 8 chunks) into LDS: 512 units of 32B,
        // each thread copies 2 contiguous units (64B) as 4x uint4.
#pragma unroll
        for (int q = 0; q < 2; ++q) {
            const int uu  = 2 * tid + q;
            const int sel = uu >> 8;
            const int ch  = (uu >> 5) & 7;
            const int ln  = uu & 31;
            const unsigned short* s =
                (sel ? fragL : fragH) + (((size_t)kt * 8 + ch) * 32 + ln) * 16;
            uint4* d = (uint4*)&Blds[sel][ch][ln][0];
            const uint4* sv = (const uint4*)s;
            d[0] = sv[0];
            d[1] = sv[1];
        }
        __syncthreads();

        v8f acc = {0.f, 0.f, 0.f, 0.f, 0.f, 0.f, 0.f, 0.f};
#pragma unroll
        for (int ch = 0; ch < 8; ++ch) {
            const v16bf bh = *(const v16bf*)&Blds[0][ch][lane][0];
            const v16bf bl = *(const v16bf*)&Blds[1][ch][lane][0];
            // dot ~= xh*eh + xh*el + xl*eh   (bf16x3 fp32 emulation)
            acc = __builtin_amdgcn_wmma_f32_16x16x32_bf16(
                false, aH[ch].v, false, bh, (short)0, acc, false, false);
            acc = __builtin_amdgcn_wmma_f32_16x16x32_bf16(
                false, aH[ch].v, false, bl, (short)0, acc, false, false);
            acc = __builtin_amdgcn_wmma_f32_16x16x32_bf16(
                false, aL[ch].v, false, bh, (short)0, acc, false, false);
        }

        // C/D layout: lane L, vgpr i -> n = L%16 (code), m = i + 8*(L/16)
        const int k  = kt * 16 + (lane & 15);
        const float e2 = enormLds[k];
#pragma unroll
        for (int i = 0; i < 8; ++i) {
            const float d = e2 - 2.0f * acc[i];
            if (d < bv[i]) { bv[i] = d; bi[i] = k; }   // k strictly increasing -> first-min kept
        }
        __syncthreads();   // protect Blds for next stage
    }

    // ---- argmin reduction across the 16 lanes holding the 16 codes ------
#pragma unroll
    for (int off = 1; off <= 8; off <<= 1) {
#pragma unroll
        for (int i = 0; i < 8; ++i) {
            const float ov = __shfl_xor(bv[i], off, 32);
            const int   oi = __shfl_xor(bi[i], off, 32);
            if (ov < bv[i] || (ov == bv[i] && oi < bi[i])) { bv[i] = ov; bi[i] = oi; }
        }
    }
    if ((lane & 15) == 0) {
        const int mbase = wave * 16 + ((lane >> 4) << 3);
#pragma unroll
        for (int i = 0; i < 8; ++i) idxLds[mbase + i] = bi[i];
    }
    __syncthreads();

    // ---- histogram (integer atomics: deterministic) ---------------------
    if (tid < 128) atomicAdd(&counts[idxLds[tid]], 1);

    // ---- out = emb[idx] (exact f32) + loss partial ----------------------
    float lacc = 0.f;
    {
        const float* inpB = inp  + (size_t)b * EMB_DIM * HW + hw0;
        float*       outB = dout + 1 + (size_t)b * EMB_DIM * HW + hw0;
        for (int id = tid; id < 128 * EMB_DIM; id += 256) {
            const int j = id & 127;        // row in tile (consecutive -> coalesced hw)
            const int c = id >> 7;
            const float x = inpB[(size_t)c * HW + j];     // L2-resident re-read
            const float e = emb[(size_t)idxLds[j] * EMB_DIM + c];
            const float df = e - x;
            lacc += df * df;
            outB[(size_t)c * HW + j] = e;
        }
    }
    red[tid] = lacc;
    __syncthreads();
#pragma unroll
    for (int s = 128; s > 0; s >>= 1) {
        if (tid < s) red[tid] += red[tid + s];
        __syncthreads();
    }
    if (tid == 0) partials[wg] = red[0];

    // ---- one-hot encodings (float2 stores; base offset is 8B-aligned) ---
    {
        float2* encB = (float2*)(dout + ENC_OFF) + (size_t)n0 * 512;
        for (int id = tid; id < 128 * 512; id += 256) {
            const int j = id >> 9;
            const int p = id & 511;
            const int idx = idxLds[j];
            const int c0 = p * 2;
            float2 v;
            v.x = (c0     == idx) ? 1.0f : 0.0f;
            v.y = (c0 + 1 == idx) ? 1.0f : 0.0f;
            encB[(size_t)j * 512 + p] = v;
        }
    }
}

// ---------------------------------------------------------------------------
// finalize: fixed-order loss sum + perplexity from histogram (deterministic)
// ---------------------------------------------------------------------------
__global__ void vq_finalize(const float* __restrict__ partials,
                            const int* __restrict__ counts,
                            float* __restrict__ dout) {
    __shared__ float red[256];
    const int tid = threadIdx.x;
    if (tid == 0) {
        float s = 0.f;
        for (int i = 0; i < 512; ++i) s += partials[i];
        dout[0] = 1.25f * s * (1.0f / (float)OUT_ELEMS);
    }
    float s = 0.f;
    for (int k = tid; k < NUM_EMB; k += 256) {
        const float p = (float)counts[k] * (1.0f / (float)NROWS);
        s += p * logf(p + 1e-10f);
    }
    red[tid] = s;
    __syncthreads();
#pragma unroll
    for (int st = 128; st > 0; st >>= 1) {
        if (tid < st) red[tid] += red[tid + st];
        __syncthreads();
    }
    if (tid == 0) dout[PERP_OFF] = expf(-red[0]);
}

// ---------------------------------------------------------------------------
extern "C" void kernel_launch(void* const* d_in, const int* in_sizes, int n_in,
                              void* d_out, int out_size, void* d_ws, size_t ws_size,
                              hipStream_t stream) {
    const float* inp = (const float*)d_in[0];   // [64,256,32,32]
    const float* emb = (const float*)d_in[1];   // [1024,256]
    float* dout = (float*)d_out;

    char* ws = (char*)d_ws;
    unsigned short* fragH   = (unsigned short*)(ws);            // 512 KB
    unsigned short* fragL   = (unsigned short*)(ws + 524288);   // 512 KB
    float*          enorm2  = (float*)(ws + 1048576);           // 4 KB
    int*            counts  = (int*)  (ws + 1052672);           // 4 KB
    float*          partials= (float*)(ws + 1056768);           // 2 KB

    vq_init    <<<1,   256, 0, stream>>>(counts);
    vq_prep_emb<<<64,  256, 0, stream>>>(emb, fragH, fragL);
    vq_enorm   <<<4,   256, 0, stream>>>(emb, enorm2);
    vq_main    <<<512, 256, 0, stream>>>(inp, emb, fragH, fragL, enorm2,
                                         counts, partials, dout);
    vq_finalize<<<1,   256, 0, stream>>>(partials, counts, dout);
}
